// MultiHeadAttention_15169824489664
// MI455X (gfx1250) — matrix operand order repaired and checked by reading, not compiled, not run on hardware
//
#include <hip/hip_runtime.h>
#include <hip/hip_bf16.h>

// ---------------------------------------------------------------------------
// MHA with heads-axis softmax quirk, bf16 WMMA GEMM pipeline for gfx1250.
//  B=8, S=1024, D=1024, H=16, E=64.
//  All GEMMs are NT (B stored [N][K] row-major) so both operands stage to LDS
//  as contiguous 16B chunks via async global->LDS copies, double-buffered,
//  with the K-loop peeled so the steady state is branch-free.
//  ws: X1b(16M) X2b(16M) Wqt/Wkt/Wvt/Wot(2M each) Qb(16M) Kb(16M) Vt(16M)
//      Ctx(16M) scores f32(512M) attn bf16(256M)  ~= 872 MB total.
// ---------------------------------------------------------------------------

typedef __bf16 v16bf __attribute__((ext_vector_type(16)));
typedef __bf16 v8bf  __attribute__((ext_vector_type(8)));
typedef float  v8f   __attribute__((ext_vector_type(8)));

static constexpr int B_ = 8;
static constexpr int S_ = 1024;
static constexpr int D_ = 1024;
static constexpr int H_ = 16;
static constexpr int E_ = 64;

static constexpr int BK   = 32;   // K-step of v_wmma_f32_16x16x32_bf16
static constexpr int LPAD = 8;    // bf16 pad per LDS row (keeps 16B alignment)
static constexpr int LSTR = BK + LPAD;

// ---------------------------------------------------------------------------
// f32 -> bf16 elementwise convert
// ---------------------------------------------------------------------------
__global__ __launch_bounds__(256)
void f32_to_bf16_kernel(const float* __restrict__ x, __bf16* __restrict__ y, int n)
{
    int i = blockIdx.x * blockDim.x + threadIdx.x;
    int stride = gridDim.x * blockDim.x;
    for (; i < n; i += stride) y[i] = (__bf16)x[i];
}

// ---------------------------------------------------------------------------
// f32 [K][N] -> bf16 [N][K] tiled transpose (for weight matrices)
// ---------------------------------------------------------------------------
__global__ __launch_bounds__(256)
void transpose_f32_to_bf16(const float* __restrict__ in, __bf16* __restrict__ out,
                           int K, int N)
{
    __shared__ __bf16 t[32][33];
    const int tx = threadIdx.x & 31;
    const int ty = threadIdx.x >> 5;          // 0..7
    const int k0 = blockIdx.y * 32;
    const int n0 = blockIdx.x * 32;
#pragma unroll
    for (int p = 0; p < 4; ++p)
        t[ty + p * 8][tx] = (__bf16)in[(long long)(k0 + ty + p * 8) * N + n0 + tx];
    __syncthreads();
#pragma unroll
    for (int p = 0; p < 4; ++p)
        out[(long long)(n0 + ty + p * 8) * K + k0 + tx] = t[tx][ty + p * 8];
}

// ---------------------------------------------------------------------------
// Softmax over the HEADS axis (torch legacy dim=1 quirk).
// One thread per (b,q,k): 16 strided reads, exp-normalize, bf16 store.
// Fully-masked (k>q) columns become uniform 1/16 exactly as the reference.
// ---------------------------------------------------------------------------
__global__ __launch_bounds__(256)
void softmax_heads_kernel(const float* __restrict__ scores, __bf16* __restrict__ attn)
{
    const long long SS  = (long long)S_ * S_;
    long long idx = (long long)blockIdx.x * blockDim.x + threadIdx.x;   // over B*S*S
    if (idx >= (long long)B_ * SS) return;
    long long b   = idx / SS;
    long long rem = idx - b * SS;

    const float* sp = scores + b * (long long)H_ * SS + rem;
    float v[H_];
    float m = -3.4e38f;
#pragma unroll
    for (int h = 0; h < H_; ++h) { v[h] = sp[(long long)h * SS]; m = fmaxf(m, v[h]); }
    float sum = 0.0f;
#pragma unroll
    for (int h = 0; h < H_; ++h) { v[h] = __expf(v[h] - m); sum += v[h]; }
    const float inv = 1.0f / sum;
    __bf16* ap = attn + b * (long long)H_ * SS + rem;
#pragma unroll
    for (int h = 0; h < H_; ++h) ap[(long long)h * SS] = (__bf16)(v[h] * inv);
}

// ---------------------------------------------------------------------------
// Async global->LDS staging of TOTAL_CHUNKS 16B chunks (rows x BK bf16 tile).
// g must point at (tile base + k0); rows laid out with stride ld.
// ---------------------------------------------------------------------------
template<int TOTAL_CHUNKS>
__device__ __forceinline__
void stage_async(const __bf16* __restrict__ g, int ld, __bf16* l, int tid)
{
    constexpr int CH = TOTAL_CHUNKS / 256;    // exact for all configs used
#pragma unroll
    for (int i = 0; i < CH; ++i) {
        const int c   = tid + i * 256;
        const int row = c >> 2;               // BK/8 == 4 chunks per row
        const int kc  = c & 3;
        const __bf16* src = g + (long long)row * ld + kc * 8;
        const unsigned dst = (unsigned)(size_t)(&l[row * LSTR + kc * 8]);
        asm volatile("global_load_async_to_lds_b128 %0, %1, off"
                     :: "v"(dst), "v"(src) : "memory");
    }
}

__device__ __forceinline__ void wait_async0()
{
    asm volatile("s_wait_asynccnt 0x0" ::: "memory");
}

// ---------------------------------------------------------------------------
// Batched NT bf16 GEMM, f32 accumulate via v_wmma_f32_16x16x32_bf16.
//   C = A(MxK) @ B(NxK)^T [+bias]; 256 threads = 8 waves per block.
//   Wave grid: (BM/WM) x (BN/WN) == 8.  Double-buffered async LDS staging,
//   last K-iteration peeled so the steady-state loop is branch-free.
//   EPI: 0 bias+f32 | 1 bias+bf16 | 2 scale+causal-mask-quirk+f32
//        3 plain bf16 | 4 bias+bf16 TRANSPOSED store (C[colg*ldc + rowg])
//   Batch z offset per operand: off = (z/BDIV)*Hi + (z%BDIV)*Lo.
//   All dims divide tiles exactly for this problem (no guards).
// ---------------------------------------------------------------------------
template<int BM, int BN, int WM, int WN, int EPI, int BDIV>
__global__ __launch_bounds__(256)
void gemm_bf16_wmma(const __bf16* __restrict__ A,
                    const __bf16* __restrict__ Bm,
                    void* __restrict__ Cm,
                    const float* __restrict__ bias,
                    int lda, int ldb, int ldc,
                    long long aHi, long long aLo,
                    long long bHi, long long bLo,
                    long long cHi, long long cLo,
                    int Kd)
{
    constexpr int WMT = WM / 16;
    constexpr int WNT = WN / 16;
    constexpr int WROWS = BM / WM;
    constexpr int ACH = (BM * BK) / 8;        // A chunks per stage
    constexpr int BCH = (BN * BK) / 8;        // B chunks per stage

    __shared__ __align__(16) __bf16 As[2][BM * LSTR];
    __shared__ __align__(16) __bf16 Bs[2][BN * LSTR];

    const int tid   = threadIdx.x;
    const int lane  = tid & 31;
    const int wave  = tid >> 5;
    const int wr    = wave % WROWS;
    const int wc    = wave / WROWS;
    const int khalf = lane >> 4;      // wave32: lanes 16-31 hold the other K half
    const int l16   = lane & 15;

    const int blockN = blockIdx.x * BN;
    const int blockM = blockIdx.y * BM;
    const int bh     = blockIdx.z;

    const long long aOff = (long long)(bh / BDIV) * aHi + (long long)(bh % BDIV) * aLo;
    const long long bOff = (long long)(bh / BDIV) * bHi + (long long)(bh % BDIV) * bLo;
    const long long cOff = (long long)(bh / BDIV) * cHi + (long long)(bh % BDIV) * cLo;

    const __bf16* Abase = A  + aOff + (long long)blockM * lda;
    const __bf16* Bbase = Bm + bOff + (long long)blockN * ldb;

    v8f acc[WMT][WNT];
#pragma unroll
    for (int mt = 0; mt < WMT; ++mt)
#pragma unroll
        for (int j = 0; j < WNT; ++j)
#pragma unroll
            for (int e = 0; e < 8; ++e) acc[mt][j][e] = 0.0f;

    // Load all fragments of one LDS buffer, then run all WMMAs back-to-back.
    auto compute_tile = [&](int cur) {
        v16bf afrag[WMT];
#pragma unroll
        for (int mt = 0; mt < WMT; ++mt) {
            // A 16x32 bf16: lane<16 -> K {0..7,16..23}; lane>=16 -> {8..15,24..31}
            const int row = wr * WM + mt * 16 + l16;
            union { v16bf f; v8bf h[2]; } u;
            u.h[0] = *(const v8bf*)(&As[cur][row * LSTR + 8 * khalf]);
            u.h[1] = *(const v8bf*)(&As[cur][row * LSTR + 16 + 8 * khalf]);
            afrag[mt] = u.f;
        }
        v16bf bfrag[WNT];
#pragma unroll
        for (int j = 0; j < WNT; ++j) {
            // B 32x16 bf16: lane<16 -> col=lane, K {0..7,16..23}; lane>=16 -> {8..15,24..31}
            const int col = wc * WN + j * 16 + l16;
            union { v16bf f; v8bf h[2]; } u;
            u.h[0] = *(const v8bf*)(&Bs[cur][col * LSTR + 8 * khalf]);
            u.h[1] = *(const v8bf*)(&Bs[cur][col * LSTR + 16 + 8 * khalf]);
            bfrag[j] = u.f;
        }
#pragma unroll
        for (int j = 0; j < WNT; ++j)
#pragma unroll
            for (int mt = 0; mt < WMT; ++mt)
                acc[mt][j] = __builtin_amdgcn_wmma_f32_16x16x32_bf16(
                    false, afrag[mt], false, bfrag[j], (short)0, acc[mt][j], false, false);
    };

    // ---- preload tile 0 -----------------------------------------------------
    stage_async<ACH>(Abase, lda, &As[0][0], tid);
    stage_async<BCH>(Bbase, ldb, &Bs[0][0], tid);
    wait_async0();
    __syncthreads();

    // ---- steady state: always stage next tile (branch-free) -----------------
    const int nIter = Kd / BK;
    for (int i = 0; i < nIter - 1; ++i) {
        const int cur = i & 1;
        stage_async<ACH>(Abase + (i + 1) * BK, lda, &As[cur ^ 1][0], tid);
        stage_async<BCH>(Bbase + (i + 1) * BK, ldb, &Bs[cur ^ 1][0], tid);
        compute_tile(cur);
        wait_async0();
        __syncthreads();
    }
    // ---- peeled last tile ---------------------------------------------------
    compute_tile((nIter - 1) & 1);

    // ---- epilogue: C/D layout VGPR r, lane<16 -> M=r, lane>=16 -> M=r+8 -----
#pragma unroll
    for (int mt = 0; mt < WMT; ++mt) {
#pragma unroll
        for (int j = 0; j < WNT; ++j) {
            const int colg = blockN + wc * WN + j * 16 + l16;
            float bval = 0.0f;
            if constexpr (EPI == 0 || EPI == 1 || EPI == 4) bval = bias[colg];
#pragma unroll
            for (int r = 0; r < 8; ++r) {
                const int rowg = blockM + wr * WM + mt * 16 + r + (khalf ? 8 : 0);
                float v = acc[mt][j][r] + bval;
                if constexpr (EPI == 0) {
                    ((float*)Cm)[cOff + (long long)rowg * ldc + colg] = v;
                } else if constexpr (EPI == 1) {
                    ((__bf16*)Cm)[cOff + (long long)rowg * ldc + colg] = (__bf16)v;
                } else if constexpr (EPI == 2) {
                    v *= 0.35355339059327373f;                 // 1/sqrt(B) quirk
                    if (colg > rowg || v == 0.0f) v = -1e9f;   // tril then ==0 -> -1e9
                    ((float*)Cm)[cOff + (long long)rowg * ldc + colg] = v;
                } else if constexpr (EPI == 3) {
                    ((__bf16*)Cm)[cOff + (long long)rowg * ldc + colg] = (__bf16)v;
                } else {  // EPI == 4: transposed bf16 store (used to build V^T)
                    ((__bf16*)Cm)[cOff + (long long)colg * ldc + rowg] = (__bf16)v;
                }
            }
        }
    }
}

// ---------------------------------------------------------------------------
extern "C" void kernel_launch(void* const* d_in, const int* in_sizes, int n_in,
                              void* d_out, int out_size, void* d_ws, size_t ws_size,
                              hipStream_t stream)
{
    const float* X1 = (const float*)d_in[0];
    const float* X2 = (const float*)d_in[1];
    const float* Wq = (const float*)d_in[2];
    const float* bq = (const float*)d_in[3];
    const float* Wk = (const float*)d_in[4];
    const float* bk = (const float*)d_in[5];
    const float* Wv = (const float*)d_in[6];
    const float* bv = (const float*)d_in[7];
    const float* Wo = (const float*)d_in[8];
    const float* bo = (const float*)d_in[9];
    float* out = (float*)d_out;

    const long long nX = (long long)B_ * S_ * D_;        // 8M
    const long long nW = (long long)D_ * D_;             // 1M
    const long long SS = (long long)S_ * S_;
    const long long nS = (long long)B_ * H_ * SS;        // 134M

    char* ws = (char*)d_ws;
    __bf16* X1b = (__bf16*)ws; ws += nX * 2;
    __bf16* X2b = (__bf16*)ws; ws += nX * 2;
    __bf16* Wqt = (__bf16*)ws; ws += nW * 2;             // [N][K] transposed
    __bf16* Wkt = (__bf16*)ws; ws += nW * 2;
    __bf16* Wvt = (__bf16*)ws; ws += nW * 2;
    __bf16* Wot = (__bf16*)ws; ws += nW * 2;
    __bf16* Qb  = (__bf16*)ws; ws += nX * 2;             // [B,S,H*E]
    __bf16* Kb  = (__bf16*)ws; ws += nX * 2;             // [B,S,H*E]
    __bf16* Vt  = (__bf16*)ws; ws += nX * 2;             // [H*E][B*S] (V^T)
    __bf16* Ctx = (__bf16*)ws; ws += nX * 2;             // [B,S,H*E]
    float*  Sc  = (float*) ws; ws += nS * 4;             // scores f32 (512 MB)
    __bf16* At  = (__bf16*)ws; ws += nS * 2;             // attn bf16 (256 MB)

    // 1) converts: activations elementwise, weights transposed to [N][K]
    f32_to_bf16_kernel<<<4096, 256, 0, stream>>>(X1, X1b, (int)nX);
    f32_to_bf16_kernel<<<4096, 256, 0, stream>>>(X2, X2b, (int)nX);
    dim3 gT(D_ / 32, D_ / 32, 1);
    transpose_f32_to_bf16<<<gT, 256, 0, stream>>>(Wq, Wqt, D_, D_);
    transpose_f32_to_bf16<<<gT, 256, 0, stream>>>(Wk, Wkt, D_, D_);
    transpose_f32_to_bf16<<<gT, 256, 0, stream>>>(Wv, Wvt, D_, D_);
    transpose_f32_to_bf16<<<gT, 256, 0, stream>>>(Wo, Wot, D_, D_);

    // 2) Q/K projections -> [B*S, D] bf16 ; V projection -> transposed V^T
    dim3 gProj(D_ / 128, (B_ * S_) / 128, 1);
    gemm_bf16_wmma<128, 128, 32, 64, 1, 1><<<gProj, 256, 0, stream>>>(
        X1b, Wqt, Qb, bq, D_, D_, D_, 0, 0, 0, 0, 0, 0, D_);
    gemm_bf16_wmma<128, 128, 32, 64, 1, 1><<<gProj, 256, 0, stream>>>(
        X2b, Wkt, Kb, bk, D_, D_, D_, 0, 0, 0, 0, 0, 0, D_);
    gemm_bf16_wmma<128, 128, 32, 64, 4, 1><<<gProj, 256, 0, stream>>>(
        X2b, Wvt, Vt, bv, D_, D_, B_ * S_, 0, 0, 0, 0, 0, 0, D_);

    // 3) scores[b,h,q,k] = (Q_h @ K_h^T) * 1/sqrt(8), mask quirk fused
    dim3 gScore(S_ / 128, S_ / 128, B_ * H_);
    gemm_bf16_wmma<128, 128, 32, 64, 2, H_><<<gScore, 256, 0, stream>>>(
        Qb, Kb, Sc, nullptr, D_, D_, S_,
        (long long)S_ * D_, (long long)E_,
        (long long)S_ * D_, (long long)E_,
        (long long)H_ * SS, SS, E_);

    // 4) softmax over heads axis
    softmax_heads_kernel<<<(unsigned)((B_ * SS) / 256), 256, 0, stream>>>(Sc, At);

    // 5) ctx[b,q,h*E+e] = attn_h @ V_h ; B operand = V^T rows [h*64+e][b*1024+k]
    dim3 gAV(E_ / 64, S_ / 128, B_ * H_);
    gemm_bf16_wmma<128, 64, 16, 64, 3, H_><<<gAV, 256, 0, stream>>>(
        At, Vt, Ctx, nullptr, S_, B_ * S_, D_,
        (long long)H_ * SS, SS,
        (long long)S_,      (long long)E_ * B_ * S_,
        (long long)S_ * D_, (long long)E_, S_);

    // 6) output projection: [8192 x 1024] @ Wo + bo -> f32 d_out
    gemm_bf16_wmma<128, 128, 32, 64, 0, 1><<<gProj, 256, 0, stream>>>(
        Ctx, Wot, out, bo, D_, D_, D_, 0, 0, 0, 0, 0, 0, D_);
}
